// PatchMLSLAttention_24919400251715
// MI455X (gfx1250) — matrix operand order repaired and checked by reading, not compiled, not run on hardware
//
#include <hip/hip_runtime.h>
#include <hip/hip_bf16.h>
#include <math.h>

// CDNA5 / gfx1250: wave32, WMMA 16x16x32 bf16 (f32 accumulate).
typedef __attribute__((ext_vector_type(16))) __bf16 v16bf;
typedef __attribute__((ext_vector_type(8)))  float  v8f;
typedef __attribute__((ext_vector_type(4)))  int    v4i;

namespace {
constexpr int kB  = 16;     // batches
constexpr int kK  = 1024;   // queries (codebook rows)
constexpr int kN  = 4096;   // keys (patch rows)
constexpr int kD  = 512;    // feature dim
constexpr int kWaves = 4;   // waves per workgroup
constexpr int kMT = 2;      // two 16-row query tiles -> 32 queries per workgroup
constexpr int kNC = 32;     // keys processed per chunk
constexpr int kDS = 128;    // d-slice per wave (4 waves * 128 = 512)
constexpr int kPad = 4;     // LDS row padding (floats) to break bank conflicts
constexpr int kChunks = kN / kNC;   // 128
}

#if __has_builtin(__builtin_amdgcn_global_load_async_to_lds_b128)
#define HAVE_ASYNC_LDS 1
#endif

__device__ __forceinline__ __bf16 to_bf16(float x) { return (__bf16)x; }

#if defined(HAVE_ASYNC_LDS)
// Async DMA of one [row of kDS floats] per lane: 32 x b128, tracked by ASYNCcnt.
typedef __attribute__((address_space(1))) v4i* gv4i_p;   // global b128
typedef __attribute__((address_space(3))) v4i* lv4i_p;   // LDS b128
__device__ __forceinline__ void async_stage_row(const float* gbase, float* lbase) {
#pragma unroll
  for (int i = 0; i < kDS / 4; ++i) {
    __builtin_amdgcn_global_load_async_to_lds_b128(
        (gv4i_p)(gbase + 4 * i), (lv4i_p)(lbase + 4 * i), 0, 0);
  }
}
__device__ __forceinline__ void wait_async0() {
#if __has_builtin(__builtin_amdgcn_s_wait_asynccnt)
  __builtin_amdgcn_s_wait_asynccnt(0);
#else
  asm volatile("s_wait_asynccnt 0x0" ::: "memory");
#endif
}
#endif

__global__ __launch_bounds__(kWaves * 32, 1)
void PatchAttn_flash_wmma(const float* __restrict__ codebook,
                          const float* __restrict__ patch,
                          float* __restrict__ out) {
  // Double-buffered per-wave staging of the K/V chunk slice (f32) + score-reduction buffer.
  __shared__ float stage[2][kWaves][kNC][kDS + kPad];  // 2*4*32*132*4 = 135.2 KB
  __shared__ float red[kWaves][32][kMT * 16];          // 16 KB

  const int lane = threadIdx.x & 31;
  const int wave = threadIdx.x >> 5;
  const int half = lane >> 4;   // lane group (0: lanes 0-15, 1: lanes 16-31)
  const int lm   = lane & 15;

  const int b     = blockIdx.x >> 5;               // 32 query tiles per batch
  const int mbase = (blockIdx.x & 31) * (16 * kMT);
  const int dw    = wave * kDS;

  // Per-lane global row base for staging (lane owns key-row `lane` of each chunk).
  const float* grow0 = patch + ((size_t)b * kN + lane) * kD + dw;

#if defined(HAVE_ASYNC_LDS)
  // Prefetch chunk 0 into buffer 0 while we load Q.
  async_stage_row(grow0, &stage[0][wave][lane][0]);
#endif

  // ---------------- Q (codebook) as WMMA-B fragments (contraction dim = d) -------------
  // B layout (16x16x32 bf16): lane -> column m = lane&15 ; e<8 -> k = 8*half + e ; e>=8 -> k = 16 + 8*half + (e-8).
  // Split f32 into bf16 hi/lo for bf16x3 score emulation; kept in VGPRs for all 128 chunks.
  v16bf qhi[kMT][4], qlo[kMT][4];
#pragma unroll
  for (int t = 0; t < kMT; ++t) {
    const float* qrow = codebook + ((size_t)b * kK + mbase + 16 * t + lm) * kD + dw;
#pragma unroll
    for (int dc = 0; dc < 4; ++dc) {
      const float* qp = qrow + 32 * dc + 8 * half;
#pragma unroll
      for (int e = 0; e < 16; ++e) {
        float  x  = qp[(e < 8) ? e : e + 8];
        __bf16 hi = to_bf16(x);
        qhi[t][dc][e] = hi;
        qlo[t][dc][e] = to_bf16(x - (float)hi);
      }
    }
  }

  const v8f vzero = {0.f, 0.f, 0.f, 0.f, 0.f, 0.f, 0.f, 0.f};
  v8f o[kMT][8];                 // O tile: rows m = v + 8*half (+16t), cols d = dw + 16j + lm
#pragma unroll
  for (int t = 0; t < kMT; ++t)
#pragma unroll
    for (int j = 0; j < 8; ++j) o[t][j] = vzero;

  float mstat[kMT], lstat[kMT];  // per-lane online-softmax state for query m = mbase+16t+lm
#pragma unroll
  for (int t = 0; t < kMT; ++t) { mstat[t] = -INFINITY; lstat[t] = 0.f; }

  for (int c = 0; c < kChunks; ++c) {
    const int buf = c & 1;

#if defined(HAVE_ASYNC_LDS)
    // Buffer `buf` was prefetched a full iteration ago; make it visible, then launch
    // the next chunk's DMA into the other buffer so it overlaps this iteration's math.
    wait_async0();
    if (c + 1 < kChunks)
      async_stage_row(grow0 + (size_t)(c + 1) * kNC * kD,
                      &stage[buf ^ 1][wave][lane][0]);
#else
    // Fallback: synchronous staging of this wave's [kNC x kDS] slice (b128 ops).
    {
      const float4* src = (const float4*)(grow0 + (size_t)c * kNC * kD);
      float4* dst = (float4*)&stage[buf][wave][lane][0];
#pragma unroll
      for (int i = 0; i < kDS / 4; ++i) dst[i] = src[i];
    }
#endif

    // ---- partial S^T = K * Q^T over this wave's d-slice (bf16x3: hh + hl + lh) ----------
    v8f sp[kMT][2];
#pragma unroll
    for (int t = 0; t < kMT; ++t) { sp[t][0] = vzero; sp[t][1] = vzero; }
#pragma unroll
    for (int nt = 0; nt < 2; ++nt) {
      // A layout (16x32 bf16): lane -> row n = 16nt + lane&15 ; e<8 -> d=32dc+8h+e ; e>=8 -> +16
      const float* prow = &stage[buf][wave][16 * nt + lm][8 * half];
#pragma unroll
      for (int dc = 0; dc < 4; ++dc) {
        const float* p0 = prow + 32 * dc;
        v16bf ahi, alo;
#pragma unroll
        for (int e = 0; e < 8; ++e) {
          float  x0 = p0[e], x1 = p0[16 + e];
          __bf16 h0 = to_bf16(x0), h1 = to_bf16(x1);
          ahi[e] = h0;                       ahi[8 + e] = h1;
          alo[e] = to_bf16(x0 - (float)h0);  alo[8 + e] = to_bf16(x1 - (float)h1);
        }
#pragma unroll
        for (int t = 0; t < kMT; ++t) {
          sp[t][nt] = __builtin_amdgcn_wmma_f32_16x16x32_bf16(
              false, ahi, false, qhi[t][dc], (short)0, sp[t][nt], false, false);
          sp[t][nt] = __builtin_amdgcn_wmma_f32_16x16x32_bf16(
              false, ahi, false, qlo[t][dc], (short)0, sp[t][nt], false, false);
          sp[t][nt] = __builtin_amdgcn_wmma_f32_16x16x32_bf16(
              false, alo, false, qhi[t][dc], (short)0, sp[t][nt], false, false);
        }
      }
    }

    // ---- cross-wave reduction of the four 128-wide d-slice partials via LDS -------------
    {
      float* w = &red[wave][lane][0];
#pragma unroll
      for (int t = 0; t < kMT; ++t)
#pragma unroll
        for (int nt = 0; nt < 2; ++nt)
#pragma unroll
          for (int v = 0; v < 8; ++v) w[t * 16 + nt * 8 + v] = sp[t][nt][v];
    }
    __syncthreads();
    float s[kMT][2][8];
#pragma unroll
    for (int t = 0; t < kMT; ++t)
#pragma unroll
      for (int nt = 0; nt < 2; ++nt)
#pragma unroll
        for (int v = 0; v < 8; ++v) s[t][nt][v] = 0.f;
    for (int w2 = 0; w2 < kWaves; ++w2) {
      const float* r = &red[w2][lane][0];
#pragma unroll
      for (int t = 0; t < kMT; ++t)
#pragma unroll
        for (int nt = 0; nt < 2; ++nt)
#pragma unroll
          for (int v = 0; v < 8; ++v) s[t][nt][v] += r[t * 16 + nt * 8 + v];
    }
    __syncthreads();   // red reusable next iteration

    // ---- online softmax (per-lane query column) + build P as A-fragment (no shuffles) ---
    // S^T C-layout: lane holds S[m = lane&15 (+16t), n = v + 8*half (+16nt)].
    // P A-fragment needs k = 8h+e (e<8) and 8h+16+(e-8) (e>=8) for m = lane&15 -> identity map.
    v16bf pa[kMT];
#pragma unroll
    for (int t = 0; t < kMT; ++t) {
      float cm = -INFINITY;
#pragma unroll
      for (int nt = 0; nt < 2; ++nt)
#pragma unroll
        for (int v = 0; v < 8; ++v) cm = fmaxf(cm, s[t][nt][v]);
      cm = fmaxf(cm, __shfl_xor(cm, 16, 32));      // combine the two lane halves
      const float mnew = fmaxf(mstat[t], cm);
      const float sc   = __expf(mstat[t] - mnew);  // first iter: exp(-inf) = 0
      mstat[t] = mnew;
      float csum = 0.f;
#pragma unroll
      for (int v = 0; v < 8; ++v) {
        float p0 = __expf(s[t][0][v] - mnew);
        float p1 = __expf(s[t][1][v] - mnew);
        csum += p0 + p1;
        pa[t][v]     = to_bf16(p0);
        pa[t][8 + v] = to_bf16(p1);
      }
      csum += __shfl_xor(csum, 16, 32);
      lstat[t] = lstat[t] * sc + csum;
      // rescale O rows: O row index is v + 8*half -> broadcast scale from owning lane
#pragma unroll
      for (int v = 0; v < 8; ++v) {
        const float srow = __shfl(sc, v + 8 * half, 32);
#pragma unroll
        for (int j = 0; j < 8; ++j) o[t][j][v] *= srow;
      }
    }

    // ---- O += P * V ; V read back from the same staged tile in B-layout -----------------
    // B layout: lane -> column d = 16j + lane&15 ; e<8 -> k(row n) = 8*half + e ; e>=8 -> 16 + 8*half + (e-8).
#pragma unroll
    for (int j = 0; j < 8; ++j) {
      v16bf vb;
#pragma unroll
      for (int e = 0; e < 16; ++e)
        vb[e] = to_bf16(stage[buf][wave][(e < 8) ? 8 * half + e : 8 + 8 * half + e][16 * j + lm]);
#pragma unroll
      for (int t = 0; t < kMT; ++t)
        o[t][j] = __builtin_amdgcn_wmma_f32_16x16x32_bf16(
            false, pa[t], false, vb, (short)0, o[t][j], false, false);
    }
  }

  // ---- normalize rows by l and store (coalesced 64B per lane half) ----------------------
#pragma unroll
  for (int t = 0; t < kMT; ++t) {
    const float linv = 1.0f / lstat[t];
#pragma unroll
    for (int v = 0; v < 8; ++v) {
      const float lb  = __shfl(linv, v + 8 * half, 32);
      const int   row = mbase + 16 * t + v + 8 * half;
      float* orow = out + ((size_t)b * kK + row) * kD + dw;
#pragma unroll
      for (int j = 0; j < 8; ++j) orow[16 * j + lm] = o[t][j][v] * lb;
    }
  }
}

extern "C" void kernel_launch(void* const* d_in, const int* in_sizes, int n_in,
                              void* d_out, int out_size, void* d_ws, size_t ws_size,
                              hipStream_t stream) {
  (void)in_sizes; (void)n_in; (void)out_size; (void)d_ws; (void)ws_size;
  const float* codebook = (const float*)d_in[0];   // [16, 1024, 512] f32
  const float* patch    = (const float*)d_in[1];   // [16, 4096, 512] f32
  float*       out      = (float*)d_out;           // [16, 1024, 512] f32

  dim3 grid(kB * (kK / (16 * kMT)));   // 16 batches * 32 query tiles = 512 workgroups
  dim3 block(kWaves * 32);             // 4 waves of 32 (wave32)
  hipLaunchKernelGGL(PatchAttn_flash_wmma, grid, block, 0, stream,
                     codebook, patch, out);
}
